// FlowGuidedDCN2_3599182594515
// MI455X (gfx1250) — matrix operand order repaired and checked by reading, not compiled, not run on hardware
//
#include <hip/hip_runtime.h>
#include <hip/hip_bf16.h>

// ---------------------------------------------------------------------------
// FlowGuidedDCN2 for gfx1250 (MI455X).
// Conv chain: implicit GEMM on v_wmma_f32_16x16x32_bf16.
//   - activations in halo-padded NHWC bf16 -> branch-free B-fragment loads
//   - packed weights staged to LDS per block via async global->LDS copies
//   - D-fragment -> 8 consecutive NHWC channels per lane
// DCN tail: gather/VALU kernel, w_dcn staged in LDS.
// ---------------------------------------------------------------------------

typedef __attribute__((ext_vector_type(16))) __bf16 v16bf;
typedef __attribute__((ext_vector_type(8)))  __bf16 v8bf;
typedef __attribute__((ext_vector_type(8)))  float  v8f;
typedef int v4i __attribute__((vector_size(16)));   // matches async builtin param

static constexpr int BN    = 8;
static constexpr int HH    = 128;
static constexpr int WW    = 128;
static constexpr int HWSZ  = HH * WW;       // 16384
static constexpr int HP    = HH + 2;        // 130 (halo)
static constexpr int WP    = WW + 2;        // 130
static constexpr int PPIX  = HP * WP;       // 16900 padded pixels / image
static constexpr int RING  = 2 * WP + 2 * HH;  // 516 halo pixels / image
static constexpr int CIN   = 64;
static constexpr int COUT  = 64;
static constexpr int GRP   = 16;
static constexpr int CFEAT = 160;           // 130 padded to 5 chunks of 32

// ---- async global->LDS helpers (gfx1250), guarded ------------------------
#if defined(__has_builtin)
#if __has_builtin(__builtin_amdgcn_global_load_async_to_lds_b128)
#define HAVE_ASYNC_LDS 1
#endif
#endif

__device__ __forceinline__ void async_wait_all() {
#if defined(__has_builtin) && __has_builtin(__builtin_amdgcn_s_wait_asynccnt)
  __builtin_amdgcn_s_wait_asynccnt(0);
#else
  asm volatile("s_wait_asynccnt 0x0" ::: "memory");
#endif
}

// copy `bytes` (multiple of 16) from global gsrc to LDS ldst, whole block
__device__ __forceinline__ void stage_to_lds(const char* gsrc, char* ldst,
                                             int bytes) {
#if defined(HAVE_ASYNC_LDS)
  for (int i = threadIdx.x * 16; i < bytes; i += 256 * 16) {
    __builtin_amdgcn_global_load_async_to_lds_b128(
        (__attribute__((address_space(1))) v4i*)(gsrc + i),
        (__attribute__((address_space(3))) v4i*)(ldst + i), 0, 0);
  }
  async_wait_all();
#else
  for (int i = threadIdx.x * 16; i < bytes; i += 256 * 16) {
    *(v8bf*)(ldst + i) = *(const v8bf*)(gsrc + i);
  }
#endif
  __syncthreads();
}

// ---------------------------------------------------------------------------
// Pack OIHW f32 weights into WMMA A-fragment order (bf16), K padded to 32.
//   co = coT*16 + (lane&15); K = (lane>>4)*16 + j
// ---------------------------------------------------------------------------
__global__ __launch_bounds__(256)
void pack_w_kernel(const float* __restrict__ w, __bf16* __restrict__ out,
                   int Cin, int chunks, int total) {
  int idx = blockIdx.x * 256 + threadIdx.x;
  if (idx >= total) return;
  int j    = idx & 15;
  int lane = (idx >> 4) & 31;
  int rest = idx >> 9;
  int chunk = rest % chunks; rest /= chunks;
  int k   = rest % 9;
  int coT = rest / 9;
  int co = coT * 16 + (lane & 15);
  int K  = ((lane >> 4) << 4) + j;
  int ci = chunk * 32 + K;
  float v = (ci < Cin) ? w[(co * Cin + ci) * 9 + k] : 0.0f;
  out[idx] = (__bf16)v;
}

// ---------------------------------------------------------------------------
// Zero the 1-pixel halo ring of a padded NHWC buffer.
// ---------------------------------------------------------------------------
__global__ __launch_bounds__(256)
void halo_zero_kernel(__bf16* __restrict__ buf, int Cpad, int total) {
  int idx = blockIdx.x * 256 + threadIdx.x;
  if (idx >= total) return;
  int c = idx % Cpad;
  int t = idx / Cpad;
  int r = t % RING;
  int b = t / RING;
  int yy, xx;
  if (r < WP)            { yy = 0;      xx = r; }
  else if (r < 2 * WP)   { yy = HP - 1; xx = r - WP; }
  else { int rr = r - 2 * WP; yy = 1 + (rr & (HH - 1)); xx = (rr >> 7) ? WP - 1 : 0; }
  buf[((size_t)(b * HP + yy) * WP + xx) * Cpad + c] = (__bf16)0.0f;
}

// ---------------------------------------------------------------------------
// feat = concat([target_f, warped_fea, flow], C) -> halo NHWC bf16 (C=160)
// ---------------------------------------------------------------------------
__global__ __launch_bounds__(256)
void build_feat_kernel(const float* __restrict__ tf, const float* __restrict__ wf,
                       const float* __restrict__ fl, __bf16* __restrict__ feat,
                       int total) {
  int idx = blockIdx.x * 256 + threadIdx.x;
  if (idx >= total) return;
  int c  = idx % CFEAT;
  int p  = idx / CFEAT;                 // b*HW + y*W + x (interior pixel)
  int b  = p >> 14;
  int hw = p & (HWSZ - 1);
  int yv = hw >> 7, xv = hw & (WW - 1);
  float v = 0.0f;
  if (c < 64)        v = tf[(b * 64 + c) * HWSZ + hw];
  else if (c < 128)  v = wf[(b * 64 + (c - 64)) * HWSZ + hw];
  else if (c == 128) v = fl[(b * 2 + 0) * HWSZ + hw];
  else if (c == 129) v = fl[(b * 2 + 1) * HWSZ + hw];
  feat[((size_t)(b * HP + yv + 1) * WP + xv + 1) * CFEAT + c] = (__bf16)v;
}

// ---------------------------------------------------------------------------
// 3x3 pad1 conv, implicit GEMM on v_wmma_f32_16x16x32_bf16.
// Block = 8 waves = one full image row (128 px) x one 16-channel co-tile.
// A-slab for this co-tile staged in LDS (async); B loads are branch-free
// thanks to the halo. Modes: 0 lrelu->bf16 halo NHWC; 1 offset f32 NHWC;
// 2 sigmoid f32 NHWC.
// ---------------------------------------------------------------------------
__global__ __launch_bounds__(256)
void conv3x3_wmma_kernel(const __bf16* __restrict__ act,
                         const __bf16* __restrict__ apack,
                         const float*  __restrict__ bias,
                         const float*  __restrict__ flow,
                         void* __restrict__ outp,
                         int Cpad, int nChunks, int nCoT, int Cout, int mode) {
  extern __shared__ __bf16 ldsA[];     // 9 * nChunks * 512 bf16

  const int coT = blockIdx.x % nCoT;
  const int pixBlock = blockIdx.x / nCoT;          // [0, 1024)
  const int lane = threadIdx.x & 31;
  const int wv = threadIdx.x >> 5;

  // stage this co-tile's packed A slab into LDS (whole block cooperates)
  const int slabBytes = 9 * nChunks * 512 * 2;
  stage_to_lds((const char*)(apack + (size_t)coT * 9 * nChunks * 512),
               (char*)ldsA, slabBytes);

  const int pix = pixBlock * 8 + wv;               // [0, 8192)
  const int xt = pix & 7;
  const int y  = (pix >> 3) & (HH - 1);
  const int b  = pix >> 10;
  const int xx = (xt << 4) + (lane & 15);          // N index (x pixel)
  const int cLaneOff = (lane >> 4) << 4;           // +16 channels hi half

  v8f acc;
  #pragma unroll
  for (int i = 0; i < 8; ++i) acc[i] = 0.0f;

  // streaming hint: next row of this image
  {
    int yp = y + 2 < HH ? y + 2 : HH - 1;
    __builtin_prefetch((const void*)(act +
        ((size_t)(b * HP + yp + 1) * WP + xx + 1) * Cpad + cLaneOff), 0, 1);
  }

  const size_t rowStride = (size_t)WP * Cpad;
  const __bf16* base = act + ((size_t)(b * HP + y) * WP + xx) * Cpad + cLaneOff;
  // base points at padded (y-1+1=y, x-1+1=x) = tap (dy=-1,dx=-1)

  for (int k = 0; k < 9; ++k) {
    const int dy = k / 3, dx = k % 3;              // 0..2 in padded coords
    const __bf16* bptr = base + (size_t)dy * rowStride + (size_t)dx * Cpad;
    const __bf16* aptr = ldsA + (k * nChunks) * 512 + lane * 16;
    for (int c = 0; c < nChunks; ++c) {
      v16bf afrag = *(const v16bf*)(aptr + c * 512);
      v16bf bfrag = *(const v16bf*)(bptr + c * 32);
      acc = __builtin_amdgcn_wmma_f32_16x16x32_bf16(
          false, afrag, false, bfrag, (short)0, acc, false, false);
    }
  }

  // D element r -> channel coT*16 + (lane>>4)*8 + r at pixel (lane&15)
  const int coBase = coT * 16 + ((lane >> 4) << 3);

  if (mode == 0) {
    __bf16* out = (__bf16*)outp;
    const size_t pb = (size_t)(b * HP + y + 1) * WP + xx + 1;  // halo interior
    v8bf r;
    #pragma unroll
    for (int i = 0; i < 8; ++i) {
      float v = acc[i] + bias[coBase + i];
      v = v > 0.0f ? v : 0.1f * v;
      r[i] = (__bf16)v;
    }
    *(v8bf*)(out + pb * Cout + coBase) = r;
  } else if (mode == 1) {
    float* out = (float*)outp;
    const size_t pb = (size_t)(b * HH + y) * WW + xx;          // flat NHWC
    const float f0 = flow[(b * 2 + 0) * HWSZ + y * WW + xx];
    const float f1 = flow[(b * 2 + 1) * HWSZ + y * WW + xx];
    v8f r;
    #pragma unroll
    for (int i = 0; i < 8; ++i) {
      int ch = coBase + i;
      float v = 20.0f * tanhf(acc[i] + bias[ch]);
      v += ((ch & 1) == 0) ? f1 : f0;              // even->flow[1], odd->flow[0]
      r[i] = v;
    }
    __builtin_nontemporal_store(r, (v8f*)(out + pb * Cout + coBase));
  } else {
    float* out = (float*)outp;
    const size_t pb = (size_t)(b * HH + y) * WW + xx;
    v8f r;
    #pragma unroll
    for (int i = 0; i < 8; ++i) {
      float v = acc[i] + bias[coBase + i];
      r[i] = 1.0f / (1.0f + expf(-v));
    }
    __builtin_nontemporal_store(r, (v8f*)(out + pb * Cout + coBase));
  }
}

// ---------------------------------------------------------------------------
// Deformable conv tail: one thread = (b, y, x, group); Og=Cg=4.
// x stays NCHW f32; w_dcn (2304 f32 = 9 KB) staged in LDS (async).
// ---------------------------------------------------------------------------
__global__ __launch_bounds__(256)
void dcn_kernel(const float* __restrict__ xin, const float* __restrict__ off,
                const float* __restrict__ msk, const float* __restrict__ wdcn,
                float* __restrict__ out) {
  __shared__ float ws[64 * 4 * 9];
  stage_to_lds((const char*)wdcn, (char*)ws, 64 * 4 * 9 * 4);

  int idx = blockIdx.x * 256 + threadIdx.x;   // B*H*W*G exact
  const int g = idx & 15;
  const int p = idx >> 4;                      // b*HW + y*W + x
  const int x = p & (WW - 1);
  const int yv = (p >> 7) & (HH - 1);
  const int b = p >> 14;

  const float* offp = off + (size_t)p * 288 + g * 18;
  const float* mskp = msk + (size_t)p * 144 + g * 9;
  const float* xg = xin + (size_t)(b * CIN + g * 4) * HWSZ;

  float acc0 = 0.f, acc1 = 0.f, acc2 = 0.f, acc3 = 0.f;

  #pragma unroll
  for (int k = 0; k < 9; ++k) {
    const float oy = __builtin_nontemporal_load(offp + 2 * k);
    const float ox = __builtin_nontemporal_load(offp + 2 * k + 1);
    const float m  = __builtin_nontemporal_load(mskp + k);
    const float ysf = (float)(yv - 1 + k / 3) + oy;
    const float xsf = (float)(x  - 1 + k % 3) + ox;
    const float y0f = floorf(ysf), x0f = floorf(xsf);
    const float wy = ysf - y0f, wx = xsf - x0f;
    const int y0 = (int)y0f, x0 = (int)x0f;

    float s0 = 0.f, s1 = 0.f, s2 = 0.f, s3 = 0.f;
    #pragma unroll
    for (int cy = 0; cy < 2; ++cy) {
      #pragma unroll
      for (int cx = 0; cx < 2; ++cx) {
        const int yi = y0 + cy, xi = x0 + cx;
        if ((unsigned)yi < (unsigned)HH && (unsigned)xi < (unsigned)WW) {
          const float wgt = (cy ? wy : 1.0f - wy) * (cx ? wx : 1.0f - wx);
          const float* src = xg + yi * WW + xi;
          s0 += wgt * src[0];
          s1 += wgt * src[HWSZ];
          s2 += wgt * src[2 * HWSZ];
          s3 += wgt * src[3 * HWSZ];
        }
      }
    }
    s0 *= m; s1 *= m; s2 *= m; s3 *= m;
    const float* wg = ws + g * 144 + k;   // ws[(g*4+og)*36 + c*9 + k]
    acc0 += wg[0]   * s0 + wg[9]   * s1 + wg[18]  * s2 + wg[27]  * s3;
    acc1 += wg[36]  * s0 + wg[45]  * s1 + wg[54]  * s2 + wg[63]  * s3;
    acc2 += wg[72]  * s0 + wg[81]  * s1 + wg[90]  * s2 + wg[99]  * s3;
    acc3 += wg[108] * s0 + wg[117] * s1 + wg[126] * s2 + wg[135] * s3;
  }

  float* op = out + (size_t)(b * COUT + g * 4) * HWSZ + yv * WW + x;
  op[0]        = acc0;
  op[HWSZ]     = acc1;
  op[2 * HWSZ] = acc2;
  op[3 * HWSZ] = acc3;
}

// ---------------------------------------------------------------------------
extern "C" void kernel_launch(void* const* d_in, const int* in_sizes, int n_in,
                              void* d_out, int out_size, void* d_ws, size_t ws_size,
                              hipStream_t stream) {
  const float* x    = (const float*)d_in[0];
  const float* wf   = (const float*)d_in[1];
  const float* tf   = (const float*)d_in[2];
  const float* fl   = (const float*)d_in[3];
  const float* ow1  = (const float*)d_in[4];  const float* ob1 = (const float*)d_in[5];
  const float* ow2  = (const float*)d_in[6];  const float* ob2 = (const float*)d_in[7];
  const float* ow3  = (const float*)d_in[8];  const float* ob3 = (const float*)d_in[9];
  const float* ow4  = (const float*)d_in[10]; const float* ob4 = (const float*)d_in[11];
  const float* mw1  = (const float*)d_in[12]; const float* mb1 = (const float*)d_in[13];
  const float* mw2  = (const float*)d_in[14]; const float* mb2 = (const float*)d_in[15];
  const float* mw3  = (const float*)d_in[16]; const float* mb3 = (const float*)d_in[17];
  const float* wdcn = (const float*)d_in[18];

  char* ws = (char*)d_ws;
  size_t off = 0;
  auto carve = [&](size_t bytes) {
    size_t r = off; off += (bytes + 255) & ~(size_t)255; return r;
  };

  const size_t nPix  = (size_t)BN * HWSZ;
  const size_t nPPix = (size_t)BN * PPIX;   // halo-padded pixels
  __bf16* feat = (__bf16*)(ws + carve(nPPix * CFEAT * 2));
  __bf16* hA   = (__bf16*)(ws + carve(nPPix * COUT * 2));
  __bf16* hB   = (__bf16*)(ws + carve(nPPix * COUT * 2));
  float*  offb = (float*) (ws + carve(nPix * 288 * 4));
  float*  mskb = (float*) (ws + carve(nPix * 144 * 4));

  // packed weight sizes: (Cout/16) * 9 * chunks * 512 bf16 elements
  const int szO1 = 4 * 9 * 5 * 512;    // Cin 130 -> 5 chunks
  const int szO2 = 4 * 9 * 2 * 512;
  const int szO4 = 18 * 9 * 2 * 512;   // Cout 288
  const int szM3 = 9 * 9 * 2 * 512;    // Cout 144
  __bf16* ow1p = (__bf16*)(ws + carve((size_t)szO1 * 2));
  __bf16* ow2p = (__bf16*)(ws + carve((size_t)szO2 * 2));
  __bf16* ow3p = (__bf16*)(ws + carve((size_t)szO2 * 2));
  __bf16* ow4p = (__bf16*)(ws + carve((size_t)szO4 * 2));
  __bf16* mw1p = (__bf16*)(ws + carve((size_t)szO1 * 2));
  __bf16* mw2p = (__bf16*)(ws + carve((size_t)szO2 * 2));
  __bf16* mw3p = (__bf16*)(ws + carve((size_t)szM3 * 2));

  auto grid = [](long long n) { return (int)((n + 255) / 256); };

  // halo zeroing (once per launch; producers fill interiors)
  halo_zero_kernel<<<grid((long long)BN * RING * CFEAT), 256, 0, stream>>>(
      feat, CFEAT, BN * RING * CFEAT);
  halo_zero_kernel<<<grid((long long)BN * RING * COUT), 256, 0, stream>>>(
      hA, COUT, BN * RING * COUT);
  halo_zero_kernel<<<grid((long long)BN * RING * COUT), 256, 0, stream>>>(
      hB, COUT, BN * RING * COUT);

  // weight packing
  pack_w_kernel<<<grid(szO1), 256, 0, stream>>>(ow1, ow1p, 130, 5, szO1);
  pack_w_kernel<<<grid(szO2), 256, 0, stream>>>(ow2, ow2p, 64, 2, szO2);
  pack_w_kernel<<<grid(szO2), 256, 0, stream>>>(ow3, ow3p, 64, 2, szO2);
  pack_w_kernel<<<grid(szO4), 256, 0, stream>>>(ow4, ow4p, 64, 2, szO4);
  pack_w_kernel<<<grid(szO1), 256, 0, stream>>>(mw1, mw1p, 130, 5, szO1);
  pack_w_kernel<<<grid(szO2), 256, 0, stream>>>(mw2, mw2p, 64, 2, szO2);
  pack_w_kernel<<<grid(szM3), 256, 0, stream>>>(mw3, mw3p, 64, 2, szM3);

  // feat concat (halo NHWC bf16, C padded to 160)
  build_feat_kernel<<<grid(nPix * CFEAT), 256, 0, stream>>>(
      tf, wf, fl, feat, (int)(nPix * CFEAT));

  // conv chain: blocks = 1024 * nCoT (blockIdx % nCoT = co-tile)
  const int smem5 = 9 * 5 * 512 * 2;   // 46080 B
  const int smem2 = 9 * 2 * 512 * 2;   // 18432 B
  conv3x3_wmma_kernel<<<1024 * 4, 256, smem5, stream>>>(
      feat, ow1p, ob1, fl, hA, CFEAT, 5, 4, 64, 0);
  conv3x3_wmma_kernel<<<1024 * 4, 256, smem2, stream>>>(
      hA, ow2p, ob2, fl, hB, 64, 2, 4, 64, 0);
  conv3x3_wmma_kernel<<<1024 * 4, 256, smem2, stream>>>(
      hB, ow3p, ob3, fl, hA, 64, 2, 4, 64, 0);
  conv3x3_wmma_kernel<<<1024 * 18, 256, smem2, stream>>>(
      hA, ow4p, ob4, fl, offb, 64, 2, 18, 288, 1);

  conv3x3_wmma_kernel<<<1024 * 4, 256, smem5, stream>>>(
      feat, mw1p, mb1, fl, hB, CFEAT, 5, 4, 64, 0);
  conv3x3_wmma_kernel<<<1024 * 4, 256, smem2, stream>>>(
      hB, mw2p, mb2, fl, hA, 64, 2, 4, 64, 0);
  conv3x3_wmma_kernel<<<1024 * 9, 256, smem2, stream>>>(
      hA, mw3p, mb3, fl, mskb, 64, 2, 9, 144, 2);

  // deformable conv tail -> d_out (NCHW f32)
  dcn_kernel<<<(BN * HWSZ * GRP) / 256, 256, 0, stream>>>(
      x, offb, mskb, wdcn, (float*)d_out);
}
